// MixingLayer_24936580121077
// MI455X (gfx1250) — matrix operand order repaired and checked by reading, not compiled, not run on hardware
//
#include <hip/hip_runtime.h>
#include <hip/hip_bf16.h>

typedef __attribute__((ext_vector_type(16))) __bf16 v16bf;
typedef __attribute__((ext_vector_type(2)))  __bf16 v2bf;
typedef __attribute__((ext_vector_type(8)))  float  v8f;
typedef __attribute__((ext_vector_type(4)))  float  v4f;

#define FDIM   128
#define L2DIM  16
#define BMTOT  1024           // b*m
#define SMP    16             // samples per workgroup
#define NSG    (BMTOT / SMP)  // 64 sample groups
#define KSPLIT 4              // K-dimension split -> 256 WGs for occupancy
#define KTOT   65536          // f*f*LMAX
#define NKT    (KTOT / 32)    // 2048 K-tiles of 32
#define NCOL   128            // 64 mx | 64 my outputs
#define WPK_BYTES ((size_t)NKT * NCOL * 64 * 2)  // 33,554,432 B (bf16 packed weights)

// LDS: Xs = f32 x-slab [s][g][c] with per-sample stride 2052 floats:
//  - rows (g,c) are 16B aligned -> ds_load b64/b128 merging in stage B
//  - bank(s) = (s*2052)%64 = 4s -> 16 samples hit disjoint bank quads (conflict-free)
#define XS_SSTRIDE 2052
#define XS_BYTES   (SMP * XS_SSTRIDE * 4)   // 131328 (32B-aligned)
#define MB_BYTES   (64 * 1024)
#define LDS_BYTES  (XS_BYTES + MB_BYTES)    // 196864 (< 320KB WGP LDS)

__device__ __forceinline__ float silu_f(float v) {
  return v * (1.0f / (1.0f + __expf(-v)));
}

// ---------------- Kernel 1: pack wx0|wy0 (f32) -> bf16 B-fragment order ----------------
// Wpk layout: per (T = global K-tile, n = output col): 64 contiguous bf16 = K 0..31 of col n.
// K order within tile T = (chunk=l*8+fblk, f', gb): kidx = f'*128 + g, K = g % 32.
__global__ __launch_bounds__(256) void pack_weights_k(const float* __restrict__ wx0,
                                                      const float* __restrict__ wy0,
                                                      v2bf* __restrict__ Wpk) {
  unsigned p = blockIdx.x * 256u + threadIdx.x;         // pair index
  if (p >= (unsigned)(NKT * NCOL * 32)) return;
  unsigned K2   = p & 31u;            // pair within tile-col
  unsigned n    = (p >> 5) & 127u;    // output column
  unsigned T    = p >> 12;            // global K-tile
  unsigned K    = K2 * 2u;
  unsigned gb   = T & 3u;
  unsigned fp   = (T >> 2) & 15u;
  unsigned chunk = T >> 6;
  unsigned fblk = chunk & 7u;
  unsigned l    = chunk >> 3;
  unsigned f    = fblk * 16u + fp;
  unsigned g0   = gb * 32u + K;
  const float* W = (n < 64u) ? wx0 : wy0;
  unsigned k = n & 63u;
  size_t base = (size_t)k * 65536u + (size_t)f * 512u + (size_t)g0 * 4u + l;
  v2bf pr;
  pr.x = (__bf16)W[base];
  pr.y = (__bf16)W[base + 4];     // g0+1 -> +4 floats (l innermost)
  Wpk[p] = pr;                    // byte addr = p*4 == ((T*128+n)*64 + K)*2
}

// ---------------- Kernel 2: fused M-formation + big bf16 WMMA GEMM ----------------
// Grid = NSG * KSPLIT workgroups. WG = 16 samples x one K-slice, 256 threads = 8 waves.
// K-slice q (<4) handles chunks {q, q+4, ..., q+28}; l = cc>>1 is compile-time after unroll.
// Wave w owns output cols [16w, 16w+16). Partial sums -> m0p[q][1024][128].
__global__ __launch_bounds__(256) void main_gemm_k(const float* __restrict__ x,
                                                   const float* __restrict__ y,
                                                   const v16bf* __restrict__ Wpk,
                                                   float* __restrict__ m0p) {
  extern __shared__ char smem[];
  float* Xs = (float*)smem;
  char*  Mb = smem + XS_BYTES;

  const int tid  = threadIdx.x;
  const int sg   = blockIdx.x & (NSG - 1);   // sample group
  const int q    = blockIdx.x >> 6;          // K-slice
  const int base = sg * SMP;

  // Cooperative float4 load of x-slab for 16 samples into LDS
  for (int i = tid; i < SMP * FDIM * L2DIM / 4; i += 256) {
    int s  = i >> 9;           // /512 float4s per sample
    int r  = i & 511;          // float4 within sample (g*4 + c4)
    v4f v = *(const v4f*)(x + (size_t)(base + s) * 2048 + r * 4);
    *(v4f*)(Xs + s * XS_SSTRIDE + r * 4) = v;
  }
  __syncthreads();

  const int s  = tid & 15;        // stage-B: sample
  const int fp = tid >> 4;        // stage-B: f' within f-block
  const int j  = tid & 31;        // stage-C: lane
  const int wave  = tid >> 5;
  const int ncol  = wave * 16 + (j & 15);
  const int khalf = j >> 4;

  v8f acc0 = {}, acc1 = {};
  const int c0tab[4] = {0, 1, 4, 9};
  const int nltab[4] = {1, 3, 5, 7};

#pragma unroll
  for (int cc = 0; cc < 8; ++cc) {
    const int chunk = q + cc * KSPLIT;
    const int l = cc >> 1;              // == chunk>>3 since q < 4 (compile-time)
    const int c0 = c0tab[l], nl = nltab[l];
    const int fblk4 = (cc & 1) * 4;     // chunk&7 == q + 4*(cc&1)

    // ---- stage B: M[s, f', g] for this (l, fblk), bf16 pre-swizzled into A-fragment layout
    {
      const int f = (fblk4 + q) * 16 + fp;
      float yv[7];
      const float* yrow = y + (size_t)(base + s) * 2048 + f * 16 + c0;
#pragma unroll
      for (int ci = 0; ci < nl; ++ci) yv[ci] = yrow[ci];
      const float* xbase = Xs + s * XS_SSTRIDE + c0;
#pragma unroll 4
      for (int gp = 0; gp < 64; ++gp) {
        const int g0 = gp * 2;
        const float* xp = xbase + g0 * 16;
        float a0 = 0.f, a1 = 0.f;
#pragma unroll
        for (int ci = 0; ci < nl; ++ci) {
          a0 = fmaf(yv[ci], xp[ci], a0);
          a1 = fmaf(yv[ci], xp[16 + ci], a1);
        }
        const int K    = g0 & 31;
        const int kt   = fp * 4 + (g0 >> 5);
        const int lane = s + ((K >> 4) << 4);
        const int pos  = K & 15;
        v2bf pr; pr.x = (__bf16)a0; pr.y = (__bf16)a1;
        *(v2bf*)(Mb + kt * 1024 + lane * 32 + pos * 2) = pr;
      }
    }
    __syncthreads();

    // ---- stage C: 64 bf16 WMMAs per wave; A register-double-buffered (DScnt<=1 waits),
    //      B streamed deep from L2-resident packed weights; dual accumulators.
    {
      const char* aptr = Mb + j * 32;
      const char* bptr = (const char*)Wpk +
          ((size_t)(chunk * 64) * 128 + ncol) * 128 + khalf * 32;
      v16bf Acur = *(const v16bf*)aptr;
#pragma unroll 8
      for (int kt = 0; kt < 64; ++kt) {
        const int ktn = (kt < 63) ? kt + 1 : 63;
        v16bf Anext = *(const v16bf*)(aptr + ktn * 1024);
        v16bf B = *(const v16bf*)(bptr + (size_t)kt * 16384);
        if (kt & 1)
          acc1 = __builtin_amdgcn_wmma_f32_16x16x32_bf16(
              false, Acur, false, B, (short)0, acc1, false, false);
        else
          acc0 = __builtin_amdgcn_wmma_f32_16x16x32_bf16(
              false, Acur, false, B, (short)0, acc0, false, false);
        Acur = Anext;
      }
    }
    __syncthreads();
  }

  // Write accumulators: C-layout VGPR r -> row M = r + 8*khalf, col = ncol
  float* mo = m0p + (size_t)q * BMTOT * 128;
#pragma unroll
  for (int r = 0; r < 8; ++r) {
    int srow = base + r + 8 * khalf;
    mo[(size_t)srow * 128 + ncol] = acc0[r] + acc1[r];
  }
}

// ---------------- Kernel 3: MLP stage (bias broadcasts over m!) ----------------
// nparts: number of K-split partial buffers to sum on input (stage 0 folds the reduction).
__global__ __launch_bounds__(256) void mlp_stage_k(const float* __restrict__ min_,
                                                   float* __restrict__ mout,
                                                   const float* __restrict__ wx,
                                                   const float* __restrict__ bx,
                                                   const float* __restrict__ wy,
                                                   const float* __restrict__ by,
                                                   int stage, int nparts) {
  int t = blockIdx.x * 256 + threadIdx.x;       // 1024*128
  if (t >= BMTOT * 128) return;
  int bm = t >> 7, col = t & 127;
  int h = col >> 6, ko = col & 63;
  int mI = bm & 63;                              // bias[stage][m][ko]
  const float* W = (h ? wy : wx) + stage * 4096 + ko;
  const float* B = (h ? by : bx) + stage * 4096 + mI * 64 + ko;
  const float* in = min_ + (size_t)bm * 128 + h * 64;
  float a = *B;
  for (int ki = 0; ki < 64; ++ki) {
    float v = in[ki];
    for (int p = 1; p < nparts; ++p) v += in[(size_t)p * BMTOT * 128 + ki];
    a = fmaf(v, W[ki * 64], a);
  }
  mout[t] = silu_f(a);
}

// ---------------- Kernel 4: final l-projection, silu gates, segment broadcast ----------------
__global__ __launch_bounds__(256) void finalize_k(const float* __restrict__ m,
                                                  const float* __restrict__ x,
                                                  const float* __restrict__ y,
                                                  const float* __restrict__ wxf,
                                                  const float* __restrict__ wyf,
                                                  float* __restrict__ out) {
  int t = blockIdx.x * 256 + threadIdx.x;       // 1024*128 = (bm, f)
  if (t >= BMTOT * FDIM) return;
  int bm = t >> 7, f = t & 127;
  const float* mx = m + (size_t)bm * 128;
  const float* my = mx + 64;
  float gx[4], gy[4];
  for (int l = 0; l < 4; ++l) {
    const float* wxp = wxf + ((size_t)l * 128 + f) * 64;
    const float* wyp = wyf + ((size_t)l * 128 + f) * 64;
    float ax = 0.f, ay = 0.f;
    for (int k = 0; k < 64; ++k) {
      ax = fmaf(mx[k], wxp[k], ax);
      ay = fmaf(my[k], wyp[k], ay);
    }
    gx[l] = silu_f(ax);
    gy[l] = silu_f(ay);
  }
  const float* xr = x + (size_t)t * 16;
  const float* yr = y + (size_t)t * 16;
  float* o = out + (size_t)t * 16;
  int c = 0;
  for (int l = 0; l < 4; ++l)
    for (int r = 0; r < 2 * l + 1; ++r, ++c)
      o[c] = gx[l] * xr[c] + gy[l] * yr[c];
}

extern "C" void kernel_launch(void* const* d_in, const int* in_sizes, int n_in,
                              void* d_out, int out_size, void* d_ws, size_t ws_size,
                              hipStream_t stream) {
  (void)in_sizes; (void)n_in; (void)out_size; (void)ws_size;
  const float* x      = (const float*)d_in[0];
  const float* y      = (const float*)d_in[1];
  const float* wx0    = (const float*)d_in[2];
  const float* wy0    = (const float*)d_in[3];
  const float* wx_mlp = (const float*)d_in[4];
  const float* bx_mlp = (const float*)d_in[5];
  const float* wy_mlp = (const float*)d_in[6];
  const float* by_mlp = (const float*)d_in[7];
  const float* wxf    = (const float*)d_in[8];
  const float* wyf    = (const float*)d_in[9];

  char* ws = (char*)d_ws;
  v2bf*  Wpk = (v2bf*)ws;                       // 33.5 MB packed bf16 weights
  float* m0p = (float*)(ws + WPK_BYTES);        // [KSPLIT][1024][128] partials
  float* m1  = m0p + (size_t)KSPLIT * BMTOT * 128;
  float* m0  = m1 + BMTOT * 128;

  // 1) weight repack: 2048*128*32 pairs
  pack_weights_k<<<(NKT * NCOL * 32) / 256, 256, 0, stream>>>(wx0, wy0, Wpk);

  // 2) fused mix-formation + WMMA GEMM, 256 WGs (64 sample groups x 4 K-slices)
  (void)hipFuncSetAttribute(reinterpret_cast<const void*>(main_gemm_k),
                            hipFuncAttributeMaxDynamicSharedMemorySize, LDS_BYTES);
  main_gemm_k<<<NSG * KSPLIT, 256, LDS_BYTES, stream>>>(x, y, (const v16bf*)Wpk, m0p);

  // 3) two MLP stages (stage 0 folds the 4-way K-split reduction into its reads)
  mlp_stage_k<<<(BMTOT * 128) / 256, 256, 0, stream>>>(m0p, m1, wx_mlp, bx_mlp, wy_mlp, by_mlp, 0, KSPLIT);
  mlp_stage_k<<<(BMTOT * 128) / 256, 256, 0, stream>>>(m1, m0, wx_mlp, bx_mlp, wy_mlp, by_mlp, 1, 1);

  // 4) gates + broadcast + combine
  finalize_k<<<(BMTOT * FDIM) / 256, 256, 0, stream>>>(m0, x, y, wxf, wyf, (float*)d_out);
}
